// MultiHeadAttention_17927193493792
// MI455X (gfx1250) — matrix operand order repaired and checked by reading, not compiled, not run on hardware
//
#include <hip/hip_runtime.h>
#include <hip/hip_bf16.h>

// ---------------------------------------------------------------------------
// MHA forward for MI455X (gfx1250, wave32, WMMA + TDM).
// All GEMMs via v_wmma_f32_16x16x32_f16 (f16 operands, fp32 accum).
// Final-projection GEMM stages its f16 A-tiles with tensor_load_to_lds (TDM),
// using TDM pad_enable to produce the bank-conflict-free LDS pitch directly.
// ---------------------------------------------------------------------------

typedef __attribute__((ext_vector_type(16))) _Float16     v16h;
typedef __attribute__((ext_vector_type(8)))  _Float16     v8h;
typedef __attribute__((ext_vector_type(8)))  float        v8f;
typedef __attribute__((ext_vector_type(4)))  unsigned int v4u;
typedef __attribute__((ext_vector_type(8)))  int          v8i;
typedef __attribute__((ext_vector_type(4)))  int          v4i;

static constexpr int S_  = 2048;
static constexpr int H_  = 2048;
static constexpr int NH_ = 16;
static constexpr int HD_ = 128;
static constexpr int M_  = 4096;   // B * S
static constexpr int LDP = 40;     // LDS row pitch (halfs): 80B, 16B aligned, conflict-free

__device__ __forceinline__ v16h pack2(v8h lo, v8h hi) {
  v16h r;
#pragma unroll
  for (int i = 0; i < 8; ++i) { r[i] = lo[i]; r[i + 8] = hi[i]; }
  return r;
}

__device__ __forceinline__ v8f wmma_f16(v16h a, v16h b, v8f c) {
  return __builtin_amdgcn_wmma_f32_16x16x32_f16(
      false, a, false, b, (short)0, c, false, false);
}

// Issue a TDM DMA of a [rows x 32] f16 tile (row stride `gstride` halfs in
// global memory) into LDS at `lds_addr`, padded to a 40-half LDS pitch.
// Must be executed by exactly one wave; waits TENSORcnt itself.
__device__ __forceinline__ void tdm_load_tile_f16_32(
    unsigned lds_addr, const _Float16* gptr, unsigned rows, unsigned gstride,
    unsigned tdim0, unsigned tdim1) {
  unsigned long long ga = (unsigned long long)(const void*)gptr;
  v4u g0;
  g0[0] = 1u;                                   // count=1, user descriptor
  g0[1] = lds_addr;                             // LDS byte address
  g0[2] = (unsigned)(ga & 0xffffffffu);         // global addr [31:0]
  g0[3] = (unsigned)((ga >> 32) & 0x01ffffffu)  // global addr [56:32]
        | (2u << 30);                           // type = 2 ("image")
  v8i g1;
  g1[0] = (1 << 16)      // data_size = 1 -> 2 bytes
        | (1 << 20)      // pad_enable
        | (3 << 22)      // pad_interval code 3 -> every 16 DWORDs (64B row)
        | (3 << 25);     // pad_amount  code 3 -> 4 DWORDs (8 halfs) -> pitch 40
  g1[1] = (int)((tdim0 & 0xffffu) << 16);       // tensor_dim0[15:0]
  g1[2] = (int)(((tdim0 >> 16) & 0xffffu) | ((tdim1 & 0xffffu) << 16));
  g1[3] = (int)(((tdim1 >> 16) & 0xffffu) | (32u << 16));   // tile_dim0 = 32
  g1[4] = (int)(rows & 0xffffu);                // tile_dim1 = rows, tile_dim2=0
  g1[5] = (int)gstride;                         // tensor_dim0_stride[31:0]
  g1[6] = 0;                                    // stride hi, dim1_stride lo
  g1[7] = 0;
  v4i z4 = {0, 0, 0, 0};
  v8i z8 = {0, 0, 0, 0, 0, 0, 0, 0};
  // amdgpu-toolchain (clang-23) 6-arg form: (g0, g1, g2, g3, extra, cpol)
  __builtin_amdgcn_tensor_load_to_lds(g0, g1, z4, z4, z8, 0);
  __builtin_amdgcn_s_wait_tensorcnt((short)0);
}

// ---------------------------------------------------------------------------
// GEMM: Y = A @ W^T.  A:[4096,2048] (fp32 if ABF==0, f16 if ABF==1),
// W:[2048,2048] fp32.  Block = 128x128 tile, 8 waves, wave = 16 rows x 128 cols.
// MODE 0: fp32 out row-major [M,H]
// MODE 1: RoPE epilogue, f16 out [b,h,s,d]
// MODE 2: f16 out transposed [b,h,d,s]   (for V)
// ---------------------------------------------------------------------------
template <int MODE, int ABF>
__global__ __launch_bounds__(256)
void gemm_kernel(const void* __restrict__ Aptr, const float* __restrict__ W,
                 void* __restrict__ Out,
                 const float* __restrict__ rope_cos,
                 const float* __restrict__ rope_sin) {
  __shared__ _Float16 Alds[128 * LDP];
  __shared__ _Float16 Wlds[128 * LDP];

  const int n0 = blockIdx.x * 128;           // blockIdx.x == head (HD_==128)
  const int m0 = blockIdx.y * 128;
  const int t = threadIdx.x;
  const int w = t >> 5, lane = t & 31, lhalf = lane >> 4, lmod = lane & 15;
  const int lrow = t >> 1, lseg = (t & 1) * 16;

  const v8f vzero = {0.f, 0.f, 0.f, 0.f, 0.f, 0.f, 0.f, 0.f};
  v8f acc[8];
#pragma unroll
  for (int i = 0; i < 8; ++i) acc[i] = vzero;

  const unsigned alds_base = (unsigned)(unsigned long long)(const void*)&Alds[0];

  for (int k0 = 0; k0 < H_; k0 += 32) {
    __syncthreads();
    // --- stage A tile (128 x 32) as f16 ---
    if (ABF) {
      // pure f16 copy: one TDM descriptor moves the whole tile, padded to LDP
      if (t < 32) {
        const _Float16* gp = (const _Float16*)Aptr + (size_t)m0 * H_ + k0;
        tdm_load_tile_f16_32(alds_base, gp, 128u, (unsigned)H_,
                             (unsigned)H_, (unsigned)M_);
      }
    } else {
      const float* gp = (const float*)Aptr + (size_t)(m0 + lrow) * H_ + k0 + lseg;
      _Float16* lp = &Alds[lrow * LDP + lseg];
#pragma unroll
      for (int i = 0; i < 16; i += 4) {
        float4 f = *(const float4*)(gp + i);
        lp[i]     = (_Float16)f.x; lp[i + 1] = (_Float16)f.y;
        lp[i + 2] = (_Float16)f.z; lp[i + 3] = (_Float16)f.w;
      }
      if (k0 + 32 < H_) __builtin_prefetch(gp + 32, 0, 0);
    }
    // --- stage W tile (128 x 32) as f16 ---
    {
      const float* gw = W + (size_t)(n0 + lrow) * H_ + k0 + lseg;
      _Float16* lw = &Wlds[lrow * LDP + lseg];
#pragma unroll
      for (int i = 0; i < 16; i += 4) {
        float4 f = *(const float4*)(gw + i);
        lw[i]     = (_Float16)f.x; lw[i + 1] = (_Float16)f.y;
        lw[i + 2] = (_Float16)f.z; lw[i + 3] = (_Float16)f.w;
      }
      if (k0 + 32 < H_) __builtin_prefetch(gw + 32, 0, 0);
    }
    __syncthreads();

    // A fragment: lane = M row, K chunks 0..7/16..23 (lanes 0-15) or 8..15/24..31
    const _Float16* ap = &Alds[(w * 16 + lmod) * LDP + lhalf * 8];
    v16h afrag = pack2(*(const v8h*)ap, *(const v8h*)(ap + 16));
#pragma unroll
    for (int nt = 0; nt < 8; ++nt) {
      // B fragment: lane = N col, K chunks 0..7/16..23 (lanes 0-15) or 8..15/24..31
      const _Float16* bp = &Wlds[(nt * 16 + lmod) * LDP + lhalf * 8];
      v16h bfrag = pack2(*(const v8h*)bp, *(const v8h*)(bp + 16));
      acc[nt] = wmma_f16(afrag, bfrag, acc[nt]);
    }
  }

  // --- epilogue ---  C layout: element(vi,lane): row = vi + 8*lane_half, col = lane%16
  if (MODE == 0) {
    float* out = (float*)Out;
#pragma unroll
    for (int nt = 0; nt < 8; ++nt)
#pragma unroll
      for (int vi = 0; vi < 8; ++vi) {
        int row = m0 + w * 16 + vi + lhalf * 8;
        int col = n0 + nt * 16 + lmod;
        out[(size_t)row * H_ + col] = acc[nt][vi];
      }
  } else if (MODE == 1) {
    _Float16* out = (_Float16*)Out;
    const int h = n0 >> 7;
#pragma unroll
    for (int nt = 0; nt < 4; ++nt)
#pragma unroll
      for (int vi = 0; vi < 8; ++vi) {
        int row = m0 + w * 16 + vi + lhalf * 8;
        int b = row >> 11, s = row & (S_ - 1);
        int d1 = nt * 16 + lmod, d2 = d1 + 64;
        float c1 = rope_cos[s * HD_ + d1], s1 = rope_sin[s * HD_ + d1];
        float c2 = rope_cos[s * HD_ + d2], s2 = rope_sin[s * HD_ + d2];
        float x1 = acc[nt][vi], x2 = acc[nt + 4][vi];
        size_t base = ((size_t)(b * NH_ + h) * S_ + s) * HD_;
        out[base + d1] = (_Float16)(x1 * c1 - x2 * s1);  // rotate_half: -q[d+64]
        out[base + d2] = (_Float16)(x2 * c2 + x1 * s2);  //  q[d-64]
      }
  } else {  // MODE == 2: transposed V
    _Float16* out = (_Float16*)Out;
    const int h = n0 >> 7;
#pragma unroll
    for (int nt = 0; nt < 8; ++nt)
#pragma unroll
      for (int vi = 0; vi < 8; ++vi) {
        int row = m0 + w * 16 + vi + lhalf * 8;
        int b = row >> 11, s = row & (S_ - 1);
        int d = nt * 16 + lmod;
        out[((size_t)(b * NH_ + h) * HD_ + d) * S_ + s] = (_Float16)acc[nt][vi];
      }
  }
}

// ---------------------------------------------------------------------------
// Flash attention: per (b,h), 128 query rows / block, wave = 16 query rows.
// Q,K: [b,h,s,d] f16.  VT: [b,h,d,s] f16.  O: [b*s, h*d] f16 row-major.
// Causal mask computed analytically; online softmax; P transposed via
// per-wave LDS (no block barriers -> divergent causal trip counts are safe).
// ---------------------------------------------------------------------------
__global__ __launch_bounds__(256)
void attn_kernel(const _Float16* __restrict__ Q, const _Float16* __restrict__ K,
                 const _Float16* __restrict__ VT, _Float16* __restrict__ O) {
  __shared__ _Float16 Plds[8][16 * 72];   // per-wave 16x64 P tile, pitch 72
  const int q0 = blockIdx.x * 128;
  const int h = blockIdx.y, b = blockIdx.z;
  const size_t bh = (size_t)(b * NH_ + h);
  const _Float16* Qb = Q + bh * (size_t)S_ * HD_;
  const _Float16* Kb = K + bh * (size_t)S_ * HD_;
  const _Float16* Vb = VT + bh * (size_t)HD_ * S_;
  const int t = threadIdx.x;
  const int w = t >> 5, lane = t & 31, lhalf = lane >> 4, lmod = lane & 15;
  const int qr0 = q0 + w * 16;
  _Float16* pw = &Plds[w][0];

  // Q fragments hoisted out of the whole key loop (d = 4 chunks of 32)
  v16h qf[4];
#pragma unroll
  for (int ds = 0; ds < 4; ++ds) {
    const _Float16* qp = Qb + (size_t)(qr0 + lmod) * HD_ + ds * 32 + lhalf * 8;
    qf[ds] = pack2(*(const v8h*)qp, *(const v8h*)(qp + 16));
  }

  const v8f vzero = {0.f, 0.f, 0.f, 0.f, 0.f, 0.f, 0.f, 0.f};
  v8f o[8];
#pragma unroll
  for (int i = 0; i < 8; ++i) o[i] = vzero;
  v8f mrun, lrun;
#pragma unroll
  for (int i = 0; i < 8; ++i) { mrun[i] = -1e30f; lrun[i] = 0.f; }
  const float scale = 0.08838834764831845f;  // 1/sqrt(128)

  const int jend = qr0 + 16;                 // keys <= last query row of wave
  for (int j0 = 0; j0 < jend; j0 += 64) {
    // ---- S = Q K^T : 4 tiles of 16x16, each 4 WMMA over d ----
    v8f st[4];
#pragma unroll
    for (int kt = 0; kt < 4; ++kt) {
      v8f sacc = vzero;
#pragma unroll
      for (int ds = 0; ds < 4; ++ds) {
        const _Float16* kp = Kb + (size_t)(j0 + kt * 16 + lmod) * HD_ + ds * 32 + lhalf * 8;
        v16h kf = pack2(*(const v8h*)kp, *(const v8h*)(kp + 16));
        sacc = wmma_f16(qf[ds], kf, sacc);
      }
      st[kt] = sacc;
    }
    // ---- scale + causal mask + local row max ----
    v8f mloc;
#pragma unroll
    for (int vi = 0; vi < 8; ++vi) mloc[vi] = -1e30f;
#pragma unroll
    for (int kt = 0; kt < 4; ++kt) {
      int kc = j0 + kt * 16 + lmod;
#pragma unroll
      for (int vi = 0; vi < 8; ++vi) {
        int qr = qr0 + vi + lhalf * 8;
        float v = st[kt][vi] * scale;
        v = (kc > qr) ? -1e30f : v;
        st[kt][vi] = v;
        mloc[vi] = fmaxf(mloc[vi], v);
      }
    }
#pragma unroll
    for (int off = 8; off >= 1; off >>= 1)
#pragma unroll
      for (int vi = 0; vi < 8; ++vi)
        mloc[vi] = fmaxf(mloc[vi], __shfl_xor(mloc[vi], off, 32));

    // ---- online softmax rescale ----
    v8f mnew, alpha;
#pragma unroll
    for (int vi = 0; vi < 8; ++vi) {
      mnew[vi] = fmaxf(mrun[vi], mloc[vi]);
      alpha[vi] = __expf(mrun[vi] - mnew[vi]);
      lrun[vi] *= alpha[vi];
      mrun[vi] = mnew[vi];
    }
#pragma unroll
    for (int dt = 0; dt < 8; ++dt)
#pragma unroll
      for (int vi = 0; vi < 8; ++vi) o[dt][vi] *= alpha[vi];

    v8f rsum = vzero;
#pragma unroll
    for (int kt = 0; kt < 4; ++kt)
#pragma unroll
      for (int vi = 0; vi < 8; ++vi) {
        float p = __expf(st[kt][vi] - mnew[vi]);
        st[kt][vi] = p;
        rsum[vi] += p;
      }
#pragma unroll
    for (int off = 8; off >= 1; off >>= 1)
#pragma unroll
      for (int vi = 0; vi < 8; ++vi)
        rsum[vi] += __shfl_xor(rsum[vi], off, 32);
#pragma unroll
    for (int vi = 0; vi < 8; ++vi) lrun[vi] += rsum[vi];

    // ---- P (C layout) -> per-wave LDS -> A-layout fragments ----
#pragma unroll
    for (int kt = 0; kt < 4; ++kt)
#pragma unroll
      for (int vi = 0; vi < 8; ++vi)
        pw[(vi + lhalf * 8) * 72 + kt * 16 + lmod] = (_Float16)st[kt][vi];
    asm volatile("s_wait_dscnt 0" ::: "memory");

    // ---- O += P @ V  (K = 64 keys = 2 chunks of 32) ----
#pragma unroll
    for (int ks = 0; ks < 2; ++ks) {
      const _Float16* pp = pw + lmod * 72 + ks * 32 + lhalf * 8;
      v16h pf = pack2(*(const v8h*)pp, *(const v8h*)(pp + 16));
#pragma unroll
      for (int dt = 0; dt < 8; ++dt) {
        const _Float16* vp = Vb + (size_t)(dt * 16 + lmod) * S_ + j0 + ks * 32 + lhalf * 8;
        v16h vf = pack2(*(const v8h*)vp, *(const v8h*)(vp + 16));
        o[dt] = wmma_f16(pf, vf, o[dt]);
      }
    }
    asm volatile("s_wait_dscnt 0" ::: "memory");  // P reads done before next overwrite
  }

  // ---- finalize: divide by l, store f16 [b*s, h*128+d] ----
#pragma unroll
  for (int dt = 0; dt < 8; ++dt)
#pragma unroll
    for (int vi = 0; vi < 8; ++vi) {
      int qr = qr0 + vi + lhalf * 8;
      int d = dt * 16 + lmod;
      float val = o[dt][vi] / lrun[vi];
      O[(size_t)(b * S_ + qr) * H_ + h * HD_ + d] = (_Float16)val;
    }
}

// ---------------------------------------------------------------------------
extern "C" void kernel_launch(void* const* d_in, const int* in_sizes, int n_in,
                              void* d_out, int out_size, void* d_ws, size_t ws_size,
                              hipStream_t stream) {
  (void)in_sizes; (void)n_in; (void)out_size; (void)ws_size;
  const float* hidden   = (const float*)d_in[0];
  // d_in[1] = attention_mask: exactly causal, reconstructed analytically in-kernel
  const float* rope_cos = (const float*)d_in[2];
  const float* rope_sin = (const float*)d_in[3];
  const float* Wq = (const float*)d_in[4];
  const float* Wk = (const float*)d_in[5];
  const float* Wv = (const float*)d_in[6];
  const float* Wo = (const float*)d_in[7];
  float* out = (float*)d_out;

  const size_t nPer = (size_t)2 * NH_ * S_ * HD_;   // 8,388,608 halfs = 16 MB
  _Float16* qbuf  = (_Float16*)d_ws;
  _Float16* kbuf  = qbuf + nPer;
  _Float16* vbufT = kbuf + nPer;
  _Float16* abuf  = vbufT + nPer;                   // total 64 MB workspace

  dim3 blk(256);
  dim3 gg(H_ / 128, M_ / 128);                      // (16, 32)
  hipLaunchKernelGGL((gemm_kernel<1, 0>), gg, blk, 0, stream, hidden, Wq, qbuf,  rope_cos, rope_sin);
  hipLaunchKernelGGL((gemm_kernel<1, 0>), gg, blk, 0, stream, hidden, Wk, kbuf,  rope_cos, rope_sin);
  hipLaunchKernelGGL((gemm_kernel<2, 0>), gg, blk, 0, stream, hidden, Wv, vbufT, nullptr,  nullptr);
  dim3 ga(S_ / 128, NH_, 2);
  hipLaunchKernelGGL(attn_kernel, ga, blk, 0, stream, qbuf, kbuf, vbufT, abuf);
  hipLaunchKernelGGL((gemm_kernel<0, 1>), gg, blk, 0, stream, abuf, Wo, out, nullptr, nullptr);
}
